// GraphEncoder_66194035966390
// MI455X (gfx1250) — matrix operand order repaired and checked by reading, not compiled, not run on hardware
//
#include <hip/hip_runtime.h>
#include <cmath>

typedef _Float16 half_t;
typedef __attribute__((ext_vector_type(16))) _Float16 v16h;
typedef __attribute__((ext_vector_type(8)))  _Float16 v8h;
typedef __attribute__((ext_vector_type(8)))  float    v8f;

#define IN_C   768
#define HID    256
#define HEADS  4
#define OUT_C  256
#define H1W    (HEADS * HID)   /* 1024 */
#define NEG_SLOPE 0.2f

// ---------------------------------------------------------------- utilities

__global__ void fill_f32(float* __restrict__ p, float v, size_t n) {
  size_t i  = (size_t)blockIdx.x * blockDim.x + threadIdx.x;
  size_t st = (size_t)gridDim.x * blockDim.x;
  for (; i < n; i += st) p[i] = v;
}

__global__ void cvt_f16(const float* __restrict__ in, half_t* __restrict__ out, size_t n) {
  size_t i  = (size_t)blockIdx.x * blockDim.x + threadIdx.x;
  size_t st = (size_t)gridDim.x * blockDim.x;
  for (; i < n; i += st) out[i] = (half_t)in[i];
}

// in[R][C] row-major  ->  out[C][R] row-major (f16)
__global__ void transpose_cvt_f16(const float* __restrict__ in, half_t* __restrict__ out,
                                  int R, int C) {
  size_t n  = (size_t)R * C;
  size_t i  = (size_t)blockIdx.x * blockDim.x + threadIdx.x;
  size_t st = (size_t)gridDim.x * blockDim.x;
  for (; i < n; i += st) {
    size_t r = i / C, c = i - r * C;
    out[c * (size_t)R + r] = (half_t)in[i];
  }
}

__device__ inline float atomicMaxFloat(float* addr, float val) {
  if (val >= 0.0f)
    return __int_as_float(atomicMax((int*)addr, __float_as_int(val)));
  else
    return __uint_as_float(atomicMin((unsigned int*)addr, __float_as_uint(val)));
}

// ---------------------------------------------------------------- WMMA GEMM
// C[M][Ncols] (f32) = A[M][K] (f16, row-major) * Bt[Ncols][K] (f16, B transposed)
// One wave32 per 16x64 output strip (1 M-tile x 4 N-tiles): the A fragment is
// loaded once per k-step and reused by 4 v_wmma_f32_16x16x32_f16 issues.
__global__ void __launch_bounds__(256) wmma_gemm_f16f32(
    const half_t* __restrict__ A, const half_t* __restrict__ Bt,
    float* __restrict__ C, int M, int Ncols, int K) {
  int wid  = blockIdx.x * (blockDim.x >> 5) + (threadIdx.x >> 5);
  int lane = threadIdx.x & 31;
  int ngrp   = Ncols >> 6;                 // groups of 4 n-tiles (Ncols % 64 == 0)
  int mtiles = M >> 4;
  if (wid >= mtiles * ngrp) return;        // wave-uniform: EXEC stays all-ones
  int tm = wid / ngrp;
  int tg = wid - tm * ngrp;
  int l15 = lane & 15;
  int hi  = lane >> 4;                     // half-wave select per ISA layouts

  const half_t* arow = A  + (size_t)(tm * 16 + l15) * K + hi * 16;
  const half_t* b0   = Bt + (size_t)(tg * 64 + l15) * K + hi * 16;
  const size_t  bstride = (size_t)16 * K;  // next n-tile's column block

  v8f acc[4] = {};
  union { v16h v; v8h h[2]; } a;
  for (int k0 = 0; k0 < K; k0 += 32) {
    // A 16x32 f16 fragment: 16 contiguous K values K=k0+16*hi+(0..15)
    a.h[0] = *(const v8h*)(arow + k0);
    a.h[1] = *(const v8h*)(arow + k0 + 8);
#pragma unroll
    for (int j = 0; j < 4; ++j) {
      // B 32x16 f16 fragment: 16 contiguous K values of column (tg*64+j*16+l15)
      v16h b = *(const v16h*)(b0 + (size_t)j * bstride + k0);
      acc[j] = __builtin_amdgcn_wmma_f32_16x16x32_f16(false, a.v, false, b,
                                                      (short)0, acc[j], false, false);
    }
  }
  // C/D layout: VGPR i -> row = i + 8*hi, col = l15
#pragma unroll
  for (int j = 0; j < 4; ++j) {
    float* crow = C + (size_t)(tm * 16 + hi * 8) * Ncols + tg * 64 + j * 16 + l15;
#pragma unroll
    for (int i = 0; i < 8; ++i) crow[(size_t)i * Ncols] = acc[j][i];
  }
}

// ------------------------------------------------------- attention score dots
// a_s[n,h] = sum_c h[n,h,c]*att_s[h,c] ; a_d likewise. One block per node.
__global__ void __launch_bounds__(256) attn_scores(
    const float* __restrict__ h, const float* __restrict__ atts,
    const float* __restrict__ attd, float* __restrict__ as_,
    float* __restrict__ ad_, int heads, int ch) {
  int node = blockIdx.x, t = threadIdx.x;
  __shared__ float ss[256], sd[256];
  const float* hrow = h + (size_t)node * heads * ch;
  for (int head = 0; head < heads; ++head) {
    float ps = 0.f, pd = 0.f;
    for (int c = t; c < ch; c += 256) {
      float v = hrow[head * ch + c];
      ps += v * atts[head * ch + c];
      pd += v * attd[head * ch + c];
    }
    ss[t] = ps; sd[t] = pd; __syncthreads();
    for (int s = 128; s > 0; s >>= 1) {
      if (t < s) { ss[t] += ss[t + s]; sd[t] += sd[t + s]; }
      __syncthreads();
    }
    if (t == 0) {
      as_[(size_t)node * heads + head] = ss[0];
      ad_[(size_t)node * heads + head] = sd[0];
    }
    __syncthreads();
  }
}

// ------------------------------------------------------------ edge softmax
__global__ void edge_e_max(const int* __restrict__ src, const int* __restrict__ dst,
                           const float* __restrict__ as_, const float* __restrict__ ad_,
                           float* __restrict__ ebuf, float* __restrict__ mmax,
                           long long EH, int heads) {
  long long i  = (long long)blockIdx.x * blockDim.x + threadIdx.x;
  long long st = (long long)gridDim.x * blockDim.x;
  for (; i < EH; i += st) {
    long long e = i / heads; int head = (int)(i - e * heads);
    int s = src[e], d = dst[e];
    float v = as_[(size_t)s * heads + head] + ad_[(size_t)d * heads + head];
    v = v > 0.f ? v : NEG_SLOPE * v;            // leaky_relu
    ebuf[i] = v;
    atomicMaxFloat(&mmax[(size_t)d * heads + head], v);
  }
}

__global__ void edge_exp_sum(const int* __restrict__ dst, float* __restrict__ ebuf,
                             const float* __restrict__ mmax, float* __restrict__ den,
                             long long EH, int heads) {
  long long i  = (long long)blockIdx.x * blockDim.x + threadIdx.x;
  long long st = (long long)gridDim.x * blockDim.x;
  for (; i < EH; i += st) {
    long long e = i / heads; int head = (int)(i - e * heads);
    int d = dst[e];
    float m = mmax[(size_t)d * heads + head];
    if (m == -INFINITY) m = 0.f;                // isolated-node fix (matches ref)
    float ex = expf(ebuf[i] - m);
    ebuf[i] = ex;
    atomicAdd(&den[(size_t)d * heads + head], ex);
  }
}

// msg = h[src]*alpha scatter-added into agg[dst]; one block per edge.
// Flattened width W = heads*ch; each thread owns 4 contiguous floats
// (single b128 gather per group; ch % 4 == 0 keeps head uniform per group).
__global__ void edge_aggregate(
    const int* __restrict__ src, const int* __restrict__ dst,
    const float* __restrict__ h, const float* __restrict__ ex,
    const float* __restrict__ den, float* __restrict__ agg, int heads, int ch) {
  int e = blockIdx.x, t = threadIdx.x;
  int s = src[e], d = dst[e];
  int W = heads * ch;
  const float* hrow = h   + (size_t)s * W;
  float*       arow = agg + (size_t)d * W;
  for (int w = t * 4; w < W; w += blockDim.x * 4) {
    int head = w / ch;
    float alpha = ex[(size_t)e * heads + head] /
                  (den[(size_t)d * heads + head] + 1e-16f);
    float4 hv = *(const float4*)(hrow + w);
    atomicAdd(arow + w + 0, hv.x * alpha);
    atomicAdd(arow + w + 1, hv.y * alpha);
    atomicAdd(arow + w + 2, hv.z * alpha);
    atomicAdd(arow + w + 3, hv.w * alpha);
  }
}

// out16 = f16(elu(agg + b))
__global__ void bias_elu_f16(const float* __restrict__ agg, const float* __restrict__ b,
                             half_t* __restrict__ out16, size_t n, int width) {
  size_t i  = (size_t)blockIdx.x * blockDim.x + threadIdx.x;
  size_t st = (size_t)gridDim.x * blockDim.x;
  for (; i < n; i += st) {
    float v = agg[i] + b[i % width];
    v = v > 0.f ? v : (expf(v) - 1.f);
    out16[i] = (half_t)v;
  }
}

__global__ void bias_add_f32(float* __restrict__ out, const float* __restrict__ b,
                             size_t n, int width) {
  size_t i  = (size_t)blockIdx.x * blockDim.x + threadIdx.x;
  size_t st = (size_t)gridDim.x * blockDim.x;
  for (; i < n; i += st) out[i] += b[i % width];
}

// ---------------------------------------------------------------- launcher

extern "C" void kernel_launch(void* const* d_in, const int* in_sizes, int n_in,
                              void* d_out, int out_size, void* d_ws, size_t ws_size,
                              hipStream_t stream) {
  (void)n_in; (void)ws_size;
  const float* x     = (const float*)d_in[0];
  const int*   ei    = (const int*)d_in[1];
  const float* W1    = (const float*)d_in[2];
  const float* atts1 = (const float*)d_in[3];
  const float* attd1 = (const float*)d_in[4];
  const float* b1    = (const float*)d_in[5];
  const float* W2    = (const float*)d_in[6];
  const float* atts2 = (const float*)d_in[7];
  const float* attd2 = (const float*)d_in[8];
  const float* b2    = (const float*)d_in[9];

  const int       N   = in_sizes[0] / IN_C;   // 10000 (= 625 * 16)
  const long long E   = in_sizes[1] / 2;      // 160000
  const int* srcv = ei;                        // edge_index[0]
  const int* dstv = ei + E;                    // edge_index[1]

  char* ws = (char*)d_ws;
  size_t off = 0;
  auto alloc = [&](size_t bytes) {
    size_t o = off; off += (bytes + 255) & ~(size_t)255; return o;
  };
  size_t o_x16  = alloc((size_t)N * IN_C * 2);
  size_t o_w1t  = alloc((size_t)H1W * IN_C * 2);
  size_t o_w2t  = alloc((size_t)OUT_C * H1W * 2);
  size_t o_h1   = alloc((size_t)N * H1W * 4);
  size_t o_as1  = alloc((size_t)N * HEADS * 4);
  size_t o_ad1  = alloc((size_t)N * HEADS * 4);
  size_t o_m1   = alloc((size_t)N * HEADS * 4);
  size_t o_dn1  = alloc((size_t)N * HEADS * 4);
  size_t o_e1   = alloc((size_t)E * HEADS * 4);
  size_t o_agg1 = alloc((size_t)N * H1W * 4);
  size_t o_h16  = alloc((size_t)N * H1W * 2);
  size_t o_as2  = alloc((size_t)N * 4);
  size_t o_ad2  = alloc((size_t)N * 4);
  size_t o_m2   = alloc((size_t)N * 4);
  size_t o_dn2  = alloc((size_t)N * 4);
  size_t o_e2   = alloc((size_t)E * 4);
  size_t o_h2   = o_x16;   // reuse: x16 dead after GEMM1; N*OUT_C*4 <= N*IN_C*2

  half_t* x16  = (half_t*)(ws + o_x16);
  half_t* w1t  = (half_t*)(ws + o_w1t);
  half_t* w2t  = (half_t*)(ws + o_w2t);
  float*  h1   = (float*)(ws + o_h1);
  float*  as1  = (float*)(ws + o_as1);
  float*  ad1  = (float*)(ws + o_ad1);
  float*  m1   = (float*)(ws + o_m1);
  float*  dn1  = (float*)(ws + o_dn1);
  float*  e1   = (float*)(ws + o_e1);
  float*  agg1 = (float*)(ws + o_agg1);
  half_t* h16  = (half_t*)(ws + o_h16);
  float*  h2   = (float*)(ws + o_h2);
  float*  as2  = (float*)(ws + o_as2);
  float*  ad2  = (float*)(ws + o_ad2);
  float*  m2   = (float*)(ws + o_m2);
  float*  dn2  = (float*)(ws + o_dn2);
  float*  e2   = (float*)(ws + o_e2);
  float*  out  = (float*)d_out;

  auto grd = [](size_t n) { return (unsigned)((n + 255) / 256); };

  // ---- precision conversion / weight transpose
  size_t nx = (size_t)N * IN_C;
  cvt_f16<<<grd(nx), 256, 0, stream>>>(x, x16, nx);
  transpose_cvt_f16<<<grd((size_t)IN_C * H1W), 256, 0, stream>>>(W1, w1t, IN_C, H1W);
  transpose_cvt_f16<<<grd((size_t)H1W * OUT_C), 256, 0, stream>>>(W2, w2t, H1W, OUT_C);

  // ---- layer 1: h1 = x @ W1  (WMMA, 16x64 strip per wave)
  int grp1 = (N / 16) * (H1W / 64);
  wmma_gemm_f16f32<<<(grp1 + 7) / 8, 256, 0, stream>>>(x16, w1t, h1, N, H1W, IN_C);

  attn_scores<<<N, 256, 0, stream>>>(h1, atts1, attd1, as1, ad1, HEADS, HID);

  fill_f32<<<grd((size_t)N * HEADS), 256, 0, stream>>>(m1, -INFINITY, (size_t)N * HEADS);
  fill_f32<<<grd((size_t)N * HEADS), 256, 0, stream>>>(dn1, 0.f, (size_t)N * HEADS);

  long long EH1 = E * HEADS;
  edge_e_max<<<grd((size_t)EH1), 256, 0, stream>>>(srcv, dstv, as1, ad1, e1, m1, EH1, HEADS);
  edge_exp_sum<<<grd((size_t)EH1), 256, 0, stream>>>(dstv, e1, m1, dn1, EH1, HEADS);

  fill_f32<<<grd((size_t)N * H1W), 256, 0, stream>>>(agg1, 0.f, (size_t)N * H1W);
  edge_aggregate<<<(unsigned)E, 256, 0, stream>>>(srcv, dstv, h1, e1, dn1, agg1, HEADS, HID);

  bias_elu_f16<<<grd((size_t)N * H1W), 256, 0, stream>>>(agg1, b1, h16, (size_t)N * H1W, H1W);

  // ---- layer 2: h2 = elu_h @ W2 (WMMA); src/dst roles swapped
  int grp2 = (N / 16) * (OUT_C / 64);
  wmma_gemm_f16f32<<<(grp2 + 7) / 8, 256, 0, stream>>>(h16, w2t, h2, N, OUT_C, H1W);

  attn_scores<<<N, 256, 0, stream>>>(h2, atts2, attd2, as2, ad2, 1, OUT_C);

  fill_f32<<<grd((size_t)N), 256, 0, stream>>>(m2, -INFINITY, (size_t)N);
  fill_f32<<<grd((size_t)N), 256, 0, stream>>>(dn2, 0.f, (size_t)N);

  // layer-2: src = edge_index[1], dst (aggregation target) = edge_index[0]
  edge_e_max<<<grd((size_t)E), 256, 0, stream>>>(dstv, srcv, as2, ad2, e2, m2, E, 1);
  edge_exp_sum<<<grd((size_t)E), 256, 0, stream>>>(srcv, e2, m2, dn2, E, 1);

  fill_f32<<<grd((size_t)out_size), 256, 0, stream>>>(out, 0.f, (size_t)out_size);
  edge_aggregate<<<(unsigned)E, 64, 0, stream>>>(dstv, srcv, h2, e2, dn2, out, 1, OUT_C);

  bias_add_f32<<<grd((size_t)out_size), 256, 0, stream>>>(out, b2, (size_t)out_size, OUT_C);
}
